// GATLayer_48524540511029
// MI455X (gfx1250) — matrix operand order repaired and checked by reading, not compiled, not run on hardware
//
#include <hip/hip_runtime.h>

typedef __attribute__((ext_vector_type(16))) __bf16 v16bf;
typedef __attribute__((ext_vector_type(8)))  float  v8f;

static __device__ __forceinline__ unsigned short f2bf(float x) {
    // round-to-nearest-even fp32 -> bf16
    unsigned int u = __float_as_uint(x);
    unsigned int r = (u + 0x7FFFu + ((u >> 16) & 1u)) >> 16;
    return (unsigned short)r;
}

union FragBF { uint4 q[2]; v16bf v; };

// ---------------------------------------------------------------------------
// prep_w: W (fp32, row-major 256 x 1024) -> bf16 B-fragments in workspace.
// 32x16 bf16 B-fragment layout used here (contiguous k, same as A):
//   frag f = ntile*8 + ktile, ntile in [0,64), ktile in [0,8)
//   lane l: column = ntile*16 + (l&15); K = v + (l<16 ? 0 : 16), v = 0..15
//   stored [f][lane][16 bf16] => 1 KB per fragment, 512 KB total.
// ---------------------------------------------------------------------------
__global__ __launch_bounds__(256) void gat_prep_w(const float* __restrict__ W,
                                                  unsigned int* __restrict__ Wb) {
    int p  = blockIdx.x * 256 + threadIdx.x;  // 0..131071, one u32 (2 bf16) each
    int u  = p & 7;
    int l  = (p >> 3) & 31;
    int f  = p >> 8;
    int kk = f & 7;
    int t  = f >> 3;
    int v  = u * 2;
    int K  = v + ((l < 16) ? 0 : 16);
    int k  = kk * 32 + K;
    int col = t * 16 + (l & 15);
    float x0 = W[(size_t)k * 1024 + col];
    float x1 = W[(size_t)(k + 1) * 1024 + col];
    Wb[p] = ((unsigned int)f2bf(x1) << 16) | (unsigned int)f2bf(x0);
}

// ---------------------------------------------------------------------------
// prep_wa: wa[hd][s][k] = sum_d W[k, hd*256+d] * a[hd, s*256+d]
// so that src[row] = h[row,:] . wa[hd][0][:], tgt likewise (exact fp32).
// ---------------------------------------------------------------------------
__global__ __launch_bounds__(256) void gat_prep_wa(const float* __restrict__ W,
                                                   const float* __restrict__ a,
                                                   float* __restrict__ wa) {
    int idx = blockIdx.x * 256 + threadIdx.x;   // 0..2047
    int k  = idx & 255;
    int s  = (idx >> 8) & 1;
    int hd = idx >> 9;
    const float* wrow = W + (size_t)k * 1024 + hd * 256;
    const float* av   = a + hd * 512 + s * 256;
    float sum = 0.0f;
    for (int d = 0; d < 256; ++d) sum = __builtin_fmaf(wrow[d], av[d], sum);
    wa[((hd * 2 + s) << 8) + k] = sum;
}

// ---------------------------------------------------------------------------
// Fused GAT kernel. One workgroup = 64 rows (32 graphs), 256 threads = 8 waves.
// A tile lives in the extended VGPR file (hoisted by the compiler from LDS);
// B fragments for each (head, n-tile) step are preloaded 8-deep so the 32
// WMMAs drain a pipelined load queue instead of blocking per fragment.
// ---------------------------------------------------------------------------
struct SMem {
    union { uint4 frag[4][8][32][2]; unsigned int w[8192]; } A;  // 32 KB A-frags
    float wav[8][256];        // wa vectors (hd*2+s)
    float red[256][8];        // per-thread dot partials
    float st[8][64];          // src/tgt per row, per (hd,s)
    float attw[4][32][2][2];  // attention weights [hd][g][i][j]
};

__global__ __launch_bounds__(256) void gat_main(const float* __restrict__ h,
                                                const int* __restrict__ adj,
                                                const uint4* __restrict__ Wb4,
                                                const float* __restrict__ wa,
                                                float* __restrict__ out) {
    __shared__ SMem sm;

    const int tid    = threadIdx.x;
    const int lane   = tid & 31;
    const int wave   = tid >> 5;            // 0..7
    const int row0   = blockIdx.x * 64;     // global row = b*2 + n
    const int graph0 = blockIdx.x * 32;
    const int l15    = lane & 15;
    const int mbase  = (lane < 16) ? 0 : 8;

    // ---- Phase 0: stage wa vectors ------------------------------------
    for (int i = tid; i < 2048; i += 256) sm.wav[i >> 8][i & 255] = wa[i];

    // ---- Phase 1: h tile (64x256 fp32) -> bf16 A-fragments in LDS -----
    // A 16x32 bf16 layout used here (contiguous k, same as B): lane<16: M=lane, K = v
    //                             lane>=16: M=lane-16, K = v + 16
    #pragma unroll
    for (int it = 0; it < 32; ++it) {
        int p  = tid + it * 256;               // 8192 bf16-pairs
        int u  = p & 7;
        int l  = (p >> 3) & 31;
        int fa = p >> 8;                       // mt*8 + kk
        int kk = fa & 7;
        int mt = fa >> 3;
        int m  = mt * 16 + (l & 15);
        int v  = u * 2;
        int K  = v + ((l < 16) ? 0 : 16);
        int k  = kk * 32 + K;
        float2 f2 = *(const float2*)(h + (size_t)(row0 + m) * 256 + k);
        sm.A.w[p] = ((unsigned int)f2bf(f2.y) << 16) | (unsigned int)f2bf(f2.x);
    }
    __syncthreads();

    // ---- Phase 2: src/tgt dots (fp32, straight from h) ----------------
    for (int pass = 0; pass < 4; ++pass) {
        int rr  = tid >> 4;                 // 0..15
        int row = pass * 16 + rr;
        const float* hrow = h + (size_t)(row0 + row) * 256;
        float acc[8];
        #pragma unroll
        for (int q = 0; q < 8; ++q) acc[q] = 0.0f;
        for (int j = 0; j < 16; ++j) {
            int c = l15 + 16 * j;
            float vv = hrow[c];
            #pragma unroll
            for (int q = 0; q < 8; ++q) acc[q] = __builtin_fmaf(vv, sm.wav[q][c], acc[q]);
        }
        #pragma unroll
        for (int q = 0; q < 8; ++q) sm.red[tid][q] = acc[q];
        __syncthreads();
        if (tid < 128) {
            int rr2 = tid >> 3, q = tid & 7;
            float s = 0.0f;
            #pragma unroll
            for (int l = 0; l < 16; ++l) s += sm.red[rr2 * 16 + l][q];
            sm.st[q][pass * 16 + rr2] = s;
        }
        __syncthreads();
    }

    // ---- Phase 3: masked leaky-relu + 2-way softmax per (head, graph) --
    if (tid < 128) {
        int g = tid & 31, hd = tid >> 5;
        const int* ad = adj + (size_t)(graph0 + g) * 4;
        float src[2], tg[2];
        src[0] = sm.st[hd * 2 + 0][2 * g];
        src[1] = sm.st[hd * 2 + 0][2 * g + 1];
        tg[0]  = sm.st[hd * 2 + 1][2 * g];
        tg[1]  = sm.st[hd * 2 + 1][2 * g + 1];
        #pragma unroll
        for (int i = 0; i < 2; ++i) {
            float e0 = src[i] + tg[0];
            float e1 = src[i] + tg[1];
            e0 = (e0 > 0.0f) ? e0 : e0 * 0.2f;
            e1 = (e1 > 0.0f) ? e1 : e1 * 0.2f;
            if (ad[i * 2 + 0] <= 0) e0 = -3.0e38f;
            if (ad[i * 2 + 1] <= 0) e1 = -3.0e38f;   // diagonal always kept
            float mx  = fmaxf(e0, e1);
            float x0  = __expf(e0 - mx);
            float x1  = __expf(e1 - mx);
            float inv = 1.0f / (x0 + x1);
            sm.attw[hd][g][i][0] = x0 * inv;
            sm.attw[hd][g][i][1] = x1 * inv;
        }
    }
    __syncthreads();

    // ---- Phase 4: GEMM + in-register attention combine -----------------
    // out[m, d] = 1/4 * sum_hd ( w[hd,g,i,0]*ht[2g, hd*256+d] + w[hd,g,i,1]*ht[2g+1, ...] )
    // Rows 2g and 2g+1 are adjacent C VGPRs in the same lane -> in-lane FMAs.
    float outacc[2][4][8];
    #pragma unroll
    for (int tt = 0; tt < 2; ++tt)
        #pragma unroll
        for (int mt = 0; mt < 4; ++mt)
            #pragma unroll
            for (int r = 0; r < 8; ++r) outacc[tt][mt][r] = 0.0f;

    for (int step = 0; step < 8; ++step) {        // step = hd*2 + tt
        const int hd = step >> 1;
        const int tt = step & 1;
        const int ntl = wave * 2 + tt;            // d-tile, identical across heads
        const uint4* bbase = Wb4 + (size_t)((hd * 16 + ntl) * 8) * 64 + lane * 2;

        if (step < 7) {
            const int ns = step + 1;
            __builtin_prefetch(
                Wb4 + (size_t)(((ns >> 1) * 16 + wave * 2 + (ns & 1)) * 8) * 64 + lane * 2, 0, 3);
        }

        // preload all 8 B fragments for this step (pipelined b128 queue)
        FragBF B[8];
        #pragma unroll
        for (int kk = 0; kk < 8; ++kk) {
            B[kk].q[0] = bbase[kk * 64];
            B[kk].q[1] = bbase[kk * 64 + 1];
        }

        v8f C[4];
        #pragma unroll
        for (int mt = 0; mt < 4; ++mt) C[mt] = (v8f){};
        #pragma unroll
        for (int kk = 0; kk < 8; ++kk) {
            #pragma unroll
            for (int mt = 0; mt < 4; ++mt) {
                FragBF af;
                af.q[0] = sm.A.frag[mt][kk][lane][0];
                af.q[1] = sm.A.frag[mt][kk][lane][1];
                C[mt] = __builtin_amdgcn_wmma_f32_16x16x32_bf16(
                            false, af.v, false, B[kk].v, (short)0, C[mt], false, false);
            }
        }

        // attention combine, in-lane: rows pair (2rp, 2rp+1) = graph gbase+rp
        #pragma unroll
        for (int mt = 0; mt < 4; ++mt) {
            const int gbase = (mt * 16 + mbase) >> 1;
            #pragma unroll
            for (int rp = 0; rp < 4; ++rp) {
                const float* wv = &sm.attw[hd][gbase + rp][0][0];
                float c0 = C[mt][2 * rp];
                float c1 = C[mt][2 * rp + 1];
                outacc[tt][mt][2 * rp]     += wv[0] * c0 + wv[1] * c1;
                outacc[tt][mt][2 * rp + 1] += wv[2] * c0 + wv[3] * c1;
            }
        }
    }

    // ---- Phase 5: mean over heads, store ------------------------------
    #pragma unroll
    for (int tt = 0; tt < 2; ++tt) {
        const int col = (wave * 2 + tt) * 16 + l15;
        #pragma unroll
        for (int mt = 0; mt < 4; ++mt) {
            #pragma unroll
            for (int r = 0; r < 8; ++r) {
                int m = mt * 16 + mbase + r;
                out[(size_t)(row0 + m) * 256 + col] = 0.25f * outacc[tt][mt][r];
            }
        }
    }
}

extern "C" void kernel_launch(void* const* d_in, const int* in_sizes, int n_in,
                              void* d_out, int out_size, void* d_ws, size_t ws_size,
                              hipStream_t stream) {
    const float* h   = (const float*)d_in[0];   // (65536, 2, 256) fp32
    const int*   adj = (const int*)d_in[1];     // (65536, 1, 2, 2) int32
    const float* W   = (const float*)d_in[2];   // (256, 1024) fp32
    const float* a   = (const float*)d_in[3];   // (4, 512, 1) fp32
    float* out = (float*)d_out;                 // (65536, 2, 256) fp32

    unsigned int* Wb = (unsigned int*)d_ws;                     // 512 KB B-frags
    float*        wa = (float*)((char*)d_ws + 512 * 1024);      // 8 KB wa vectors

    gat_prep_w <<<512, 256, 0, stream>>>(W, Wb);
    gat_prep_wa<<<8,   256, 0, stream>>>(W, a, wa);
    gat_main   <<<2048, 256, 0, stream>>>(h, adj, (const uint4*)d_ws, wa, out);
}
